// Model_78726750535936
// MI455X (gfx1250) — matrix operand order repaired and checked by reading, not compiled, not run on hardware
//
#include <hip/hip_runtime.h>
#include <hip/hip_bf16.h>
#include <math.h>

// ---------------- problem constants ----------------
#define B_   128
#define S_   256
#define E_   512
#define H_   512
#define V_   32000

typedef __bf16 bf16;
typedef __attribute__((ext_vector_type(16))) __bf16 v16bf;
typedef __attribute__((ext_vector_type(8)))  float  v8f;

union FragAB { v16bf v; uint4 q[2]; };

__device__ __forceinline__ float sigmoidf_(float x) { return 1.0f / (1.0f + __expf(-x)); }

// A-tile LDS: 16 rows x (1024 + 8 pad) bf16  -> row stride 2064B = 4-bank shift per row
#define LDS_LDA 1032

// ---------------------------------------------------------------------------
// Pack fp32 weight W[N][ldw] (rows = output cols, cols = K) into bf16 WMMA
// B-fragment layout: dst[((jt*KCtot + kc0+kc)*32 + lane)*16 + e]
//   = W[jt*16 + lane%16][coloff + kc*32 + (lane/16)*16 + e]
// ---------------------------------------------------------------------------
__global__ void pack_w_kernel(const float* __restrict__ W, int ldw, int coloff,
                              int KCsrc, int kc0, int KCtot, bf16* __restrict__ dst)
{
    int tid  = threadIdx.x;
    int lane = tid & 31, sub = tid >> 5;
    int jt = blockIdx.x;
    int kc = blockIdx.y * 8 + sub;
    if (kc >= KCsrc) return;
    int n     = jt * 16 + (lane & 15);
    int kbase = coloff + kc * 32 + (lane >> 4) * 16;
    const float* src = W + (long)n * ldw + kbase;
    bf16* d = dst + (((long)(jt * KCtot + kc0 + kc)) * 32 + lane) * 16;
#pragma unroll
    for (int e = 0; e < 16; ++e) d[e] = (bf16)src[e];
}

// Embedding gathers -> bf16 row-major
__global__ void embed_src_kernel(const int* __restrict__ tok, const float* __restrict__ emb,
                                 bf16* __restrict__ xbf)
{
    int sb = blockIdx.x;                 // sb = s*B + b
    int s = sb >> 7, b = sb & 127;
    int t = tok[b * S_ + s];             // src_tokens is [B][S]
    const float* src = emb + (long)t * E_;
    bf16* dst = xbf + (long)sb * E_;
    for (int c = threadIdx.x; c < E_; c += blockDim.x) dst[c] = (bf16)src[c];
}

__global__ void embed_trg_kernel(const int* __restrict__ tok, const float* __restrict__ emb,
                                 bf16* __restrict__ out)
{
    int b = blockIdx.x;
    int t = tok[b];
    const float* src = emb + (long)t * E_;
    bf16* dst = out + (long)b * E_;
    for (int c = threadIdx.x; c < E_; c += blockDim.x) dst[c] = (bf16)src[c];
}

__global__ void zero_kernel(float* p, int n)
{
    int i = blockIdx.x * blockDim.x + threadIdx.x;
    if (i < n) p[i] = 0.f;
}

__global__ void tanh_kernel(const float* __restrict__ src, float* __restrict__ dst, int n)
{
    int i = blockIdx.x * blockDim.x + threadIdx.x;
    if (i < n) dst[i] = tanhf(src[i]);
}

// ---------------------------------------------------------------------------
// Fused LSTM step:  g = [x_t | h_prev] @ W'^T + bih + bhh ; gates ; mask
// grid = (H/64, B/16), block = 128 (4 waves). Wave w owns h-col tile
// jc = bx*4+w and computes 4 gate tiles (i,f,g,o) with one shared A-tile.
// ---------------------------------------------------------------------------
__global__ __launch_bounds__(128)
void lstm_step_kernel(const bf16* __restrict__ xrow,    // [B][512] bf16, timestep slice
                      const float* __restrict__ hprev,  // [B][512] f32
                      float* __restrict__ hout,         // [B][512] f32
                      float* __restrict__ cbuf,         // [B][ldc] f32 in/out (owner-only)
                      int ldc,
                      const bf16* __restrict__ bpack,   // [128][32][32][16] bf16
                      const float* __restrict__ bih, const float* __restrict__ bhh,
                      const int* __restrict__ lens, int t,
                      bf16*  __restrict__ ybf,          // optional: [B][512] bf16 (t slice)
                      float* __restrict__ ysf,          // optional: src_states [B][S][H]
                      float* __restrict__ hout2, int ldh2) // optional extra h copy
{
    __shared__ bf16 As[16 * LDS_LDA];

    int tid = threadIdx.x;
    int bi  = blockIdx.y;

    // stage A-tile: cols 0..511 = x_t (already bf16), cols 512..1023 = cvt(h_prev)
    const bf16* xg = xrow + (long)(bi * 16) * 512;
    for (int idx = tid; idx < 2048; idx += 128) {
        int r = idx >> 7, c4 = (idx & 127) << 2;
        *(uint2*)&As[r * LDS_LDA + c4] = *(const uint2*)&xg[r * 512 + c4];
    }
    const float* hg = hprev + (long)(bi * 16) * 512;
    for (int idx = tid; idx < 2048; idx += 128) {
        int r = idx >> 7, c4 = (idx & 127) << 2;
        float4 f = *(const float4*)&hg[r * 512 + c4];
        bf16* d = &As[r * LDS_LDA + 512 + c4];
        d[0] = (bf16)f.x; d[1] = (bf16)f.y; d[2] = (bf16)f.z; d[3] = (bf16)f.w;
    }
    __syncthreads();

    int wave = tid >> 5, lane = tid & 31;
    int half = lane >> 4, n = lane & 15;
    int jc = blockIdx.x * 4 + wave;          // h-space tile 0..31

    v8f acc[4] = {};
    const bf16* bp[4];
#pragma unroll
    for (int q = 0; q < 4; ++q)              // gate q tile index jt = q*32 + jc
        bp[q] = bpack + (((long)(q * 32 + jc)) * 32 * 32 + lane) * 16;

    for (int kc = 0; kc < 32; ++kc) {
        FragAB a;
        const bf16* ar = &As[n * LDS_LDA + kc * 32 + half * 16];
        a.q[0] = *(const uint4*)ar;
        a.q[1] = *(const uint4*)(ar + 8);
#pragma unroll
        for (int q = 0; q < 4; ++q) {
            FragAB bb;
            const bf16* bq = bp[q] + (long)kc * 512;
            bb.q[0] = *(const uint4*)bq;
            bb.q[1] = *(const uint4*)(bq + 8);
            acc[q] = __builtin_amdgcn_wmma_f32_16x16x32_bf16(
                false, a.v, false, bb.v, (short)0, acc[q], false, false);
        }
    }

    // gate math + masking
    int colh = jc * 16 + n;                  // 0..511
    float b0 = bih[colh]          + bhh[colh];
    float b1 = bih[colh + 512]    + bhh[colh + 512];
    float b2 = bih[colh + 1024]   + bhh[colh + 1024];
    float b3 = bih[colh + 1536]   + bhh[colh + 1536];
#pragma unroll
    for (int r = 0; r < 8; ++r) {
        int b = bi * 16 + r + half * 8;      // C/D layout: M = r + 8*half
        float gi = sigmoidf_(acc[0][r] + b0);
        float gf = sigmoidf_(acc[1][r] + b1);
        float gg = tanhf    (acc[2][r] + b2);
        float go = sigmoidf_(acc[3][r] + b3);
        float cp = cbuf[(long)b * ldc + colh];
        float cn = gf * cp + gi * gg;
        float hn = go * tanhf(cn);
        bool  m  = lens ? (t < lens[b]) : true;
        float hp = hprev[(long)b * 512 + colh];
        float h2 = m ? hn : hp;
        float c2 = m ? cn : cp;
        hout[(long)b * 512 + colh] = h2;
        cbuf[(long)b * ldc + colh] = c2;
        if (ybf)   ybf[(long)b * 512 + colh] = (bf16)(m ? hn : 0.f);
        if (ysf)   ysf[((long)b * S_ + t) * 512 + colh] = m ? hn : 0.f;
        if (hout2) hout2[(long)b * ldh2 + colh] = h2;
    }
}

// ---------------------------------------------------------------------------
// Generic GEMM: out[M][N] = cvt_bf16(A f32 [M][K]) @ Bpack + bias
// grid = (N/64, M/16), block = 128 (4 waves, 16x16 tile each)
// ---------------------------------------------------------------------------
__global__ __launch_bounds__(128)
void gemm_kernel(const float* __restrict__ A, int K,
                 const bf16* __restrict__ bpack, int KCtot,
                 const float* __restrict__ bias,
                 float* __restrict__ out, int ldo)
{
    __shared__ bf16 As[16 * LDS_LDA];
    int tid = threadIdx.x;
    int bi  = blockIdx.y;

    const float* ag = A + (long)(bi * 16) * K;
    int kq = K >> 2;
    for (int idx = tid; idx < 16 * kq; idx += 128) {
        int r = idx / kq, c4 = (idx % kq) << 2;
        float4 f = *(const float4*)&ag[(long)r * K + c4];
        bf16* d = &As[r * LDS_LDA + c4];
        d[0] = (bf16)f.x; d[1] = (bf16)f.y; d[2] = (bf16)f.z; d[3] = (bf16)f.w;
    }
    __syncthreads();

    int wave = tid >> 5, lane = tid & 31;
    int half = lane >> 4, n = lane & 15;
    int jt = blockIdx.x * 4 + wave;

    v8f acc = {};
    const bf16* bp = bpack + (((long)jt * KCtot) * 32 + lane) * 16;
    for (int kc = 0; kc < KCtot; ++kc) {
        FragAB a, bb;
        const bf16* ar = &As[n * LDS_LDA + kc * 32 + half * 16];
        a.q[0] = *(const uint4*)ar;
        a.q[1] = *(const uint4*)(ar + 8);
        const bf16* bq = bp + (long)kc * 512;
        bb.q[0] = *(const uint4*)bq;
        bb.q[1] = *(const uint4*)(bq + 8);
        acc = __builtin_amdgcn_wmma_f32_16x16x32_bf16(
            false, a.v, false, bb.v, (short)0, acc, false, false);
    }

    int col = jt * 16 + n;
    float bv = bias ? bias[col] : 0.f;
#pragma unroll
    for (int r = 0; r < 8; ++r) {
        int m = bi * 16 + r + half * 8;
        out[(long)m * ldo + col] = acc[r] + bv;
    }
}

// ---------------------------------------------------------------------------
// Attention: scores/softmax/context per batch row (33 MFLOP -> VALU fine)
// ---------------------------------------------------------------------------
__global__ __launch_bounds__(256)
void attention_kernel(const float* __restrict__ ysf,  // [B][S][H]
                      const float* __restrict__ h,    // [B][H]
                      const int*   __restrict__ lens,
                      float* __restrict__ ctx_out,    // [B][H]  (d_out)
                      float* __restrict__ acat)       // [B][1024], cols 512.. = context
{
    __shared__ float hsh[512];
    __shared__ float sc[256];
    __shared__ float red[8];

    int b = blockIdx.x, tid = threadIdx.x;
    int wave = tid >> 5, lane = tid & 31;
    int len = lens[b];

    for (int c = tid; c < 512; c += 256) hsh[c] = h[(long)b * 512 + c];
    __syncthreads();

    for (int s = wave; s < 256; s += 8) {
        const float* row = ysf + ((long)b * 256 + s) * 512;
        float p = 0.f;
        for (int k = lane; k < 512; k += 32) p += row[k] * hsh[k];
#pragma unroll
        for (int off = 16; off > 0; off >>= 1) p += __shfl_xor(p, off, 32);
        if (lane == 0) sc[s] = (s < len) ? p : -__builtin_inff();
    }
    __syncthreads();

    float v = sc[tid];
    float mx = v;
#pragma unroll
    for (int off = 16; off > 0; off >>= 1) mx = fmaxf(mx, __shfl_xor(mx, off, 32));
    if (lane == 0) red[wave] = mx;
    __syncthreads();
    float bmx = red[0];
#pragma unroll
    for (int i = 1; i < 8; ++i) bmx = fmaxf(bmx, red[i]);
    float ex = __expf(v - bmx);
    float sm = ex;
#pragma unroll
    for (int off = 16; off > 0; off >>= 1) sm += __shfl_xor(sm, off, 32);
    __syncthreads();
    if (lane == 0) red[wave] = sm;
    __syncthreads();
    float bsm = 0.f;
#pragma unroll
    for (int i = 0; i < 8; ++i) bsm += red[i];
    sc[tid] = ex / bsm;
    __syncthreads();

    for (int col = tid; col < 512; col += 256) {
        float a = 0.f;
        for (int s = 0; s < 256; ++s) a += sc[s] * ysf[((long)b * 256 + s) * 512 + col];
        ctx_out[(long)b * 512 + col] = a;
        acat[(long)b * 1024 + 512 + col] = a;
    }
}

__global__ __launch_bounds__(256)
void argmax_kernel(const float* __restrict__ logits, float* __restrict__ out)
{
    __shared__ float vals[256];
    __shared__ int   idxs[256];
    int b = blockIdx.x, tid = threadIdx.x;
    const float* row = logits + (long)b * V_;
    float bv = -__builtin_inff();
    int bidx = 0;
    for (int j = tid; j < V_; j += 256) {
        float x = row[j];
        if (x > bv) { bv = x; bidx = j; }
    }
    vals[tid] = bv; idxs[tid] = bidx;
    __syncthreads();
    for (int off = 128; off > 0; off >>= 1) {
        if (tid < off) {
            float ov = vals[tid + off]; int oi = idxs[tid + off];
            if (ov > vals[tid] || (ov == vals[tid] && oi < idxs[tid])) {
                vals[tid] = ov; idxs[tid] = oi;
            }
        }
        __syncthreads();
    }
    if (tid == 0) out[b] = (float)idxs[0];
}

// ---------------------------------------------------------------------------
extern "C" void kernel_launch(void* const* d_in, const int* in_sizes, int n_in,
                              void* d_out, int out_size, void* d_ws, size_t ws_size,
                              hipStream_t stream)
{
    (void)in_sizes; (void)n_in; (void)out_size; (void)ws_size;

    const int*   src_tokens = (const int*)  d_in[0];
    const int*   src_lens   = (const int*)  d_in[1];
    const int*   trg_tokens = (const int*)  d_in[2];
    const float* src_emb    = (const float*)d_in[3];
    const float* trg_emb    = (const float*)d_in[4];
    const float* enc_Wih0   = (const float*)d_in[5];
    const float* enc_Whh0   = (const float*)d_in[6];
    const float* enc_bih0   = (const float*)d_in[7];
    const float* enc_bhh0   = (const float*)d_in[8];
    const float* enc_Wih1   = (const float*)d_in[9];
    const float* enc_Whh1   = (const float*)d_in[10];
    const float* enc_bih1   = (const float*)d_in[11];
    const float* enc_bhh1   = (const float*)d_in[12];
    const float* dec_Wih    = (const float*)d_in[13];
    const float* dec_Whh    = (const float*)d_in[14];
    const float* dec_bih    = (const float*)d_in[15];
    const float* dec_bhh    = (const float*)d_in[16];
    const float* W_init     = (const float*)d_in[17];
    const float* b_init     = (const float*)d_in[18];
    const float* W_hid      = (const float*)d_in[19];
    const float* b_hid      = (const float*)d_in[20];
    const float* W_out      = (const float*)d_in[21];
    const float* b_out      = (const float*)d_in[22];
    float* dout = (float*)d_out;

    uint8_t* ws = (uint8_t*)d_ws;
    size_t off = 0;
    auto wsa = [&](size_t bytes) -> void* {
        void* p = ws + off;
        off = (off + bytes + 255) & ~((size_t)255);
        return p;
    };

    bf16*  XBF     = (bf16*) wsa((size_t)S_ * B_ * E_ * 2);   // 32 MB
    bf16*  YBF     = (bf16*) wsa((size_t)S_ * B_ * H_ * 2);   // 32 MB
    float* YSF     = (float*)wsa((size_t)B_ * S_ * H_ * 4);   // 64 MB src_states
    bf16*  packE0  = (bf16*) wsa((size_t)2048 * 1024 * 2);
    bf16*  packE1  = (bf16*) wsa((size_t)2048 * 1024 * 2);
    bf16*  packD   = (bf16*) wsa((size_t)2048 * 1024 * 2);
    bf16*  packI   = (bf16*) wsa((size_t)512 * 1024 * 2);
    bf16*  packHid = (bf16*) wsa((size_t)512 * 1024 * 2);
    bf16*  packOut = (bf16*) wsa((size_t)V_ * 512 * 2);       // 32 MB
    float* HA      = (float*)wsa((size_t)B_ * H_ * 4);
    float* HB      = (float*)wsa((size_t)B_ * H_ * 4);
    float* FIN     = (float*)wsa((size_t)B_ * 1024 * 4);      // [c0f | c1f]
    float* CDEC    = (float*)wsa((size_t)B_ * H_ * 4);
    float* HDEC    = (float*)wsa((size_t)B_ * H_ * 4);
    bf16*  EMBT    = (bf16*) wsa((size_t)B_ * E_ * 2);
    float* ACAT    = (float*)wsa((size_t)B_ * 1024 * 4);      // [h | context]
    float* ATTNV   = (float*)wsa((size_t)B_ * H_ * 4);
    float* LOGITS  = (float*)wsa((size_t)B_ * V_ * 4);        // 16 MB

    // ---- one-time weight packing (B-fragment bf16 layout) ----
    pack_w_kernel<<<dim3(128, 2), 256, 0, stream>>>(enc_Wih0, 512, 0, 16, 0, 32, packE0);
    pack_w_kernel<<<dim3(128, 2), 256, 0, stream>>>(enc_Whh0, 512, 0, 16, 16, 32, packE0);
    pack_w_kernel<<<dim3(128, 2), 256, 0, stream>>>(enc_Wih1, 512, 0, 16, 0, 32, packE1);
    pack_w_kernel<<<dim3(128, 2), 256, 0, stream>>>(enc_Whh1, 512, 0, 16, 16, 32, packE1);
    pack_w_kernel<<<dim3(128, 2), 256, 0, stream>>>(dec_Wih, 1024, 0, 16, 0, 32, packD);
    pack_w_kernel<<<dim3(128, 2), 256, 0, stream>>>(dec_Whh, 512, 0, 16, 16, 32, packD);
    pack_w_kernel<<<dim3(32, 4),  256, 0, stream>>>(W_init, 1024, 0, 32, 0, 32, packI);
    pack_w_kernel<<<dim3(32, 4),  256, 0, stream>>>(W_hid,  1024, 0, 32, 0, 32, packHid);
    pack_w_kernel<<<dim3(2000, 2),256, 0, stream>>>(W_out,   512, 0, 16, 0, 16, packOut);

    // ---- embeddings ----
    embed_src_kernel<<<S_ * B_, 256, 0, stream>>>(src_tokens, src_emb, XBF);
    embed_trg_kernel<<<B_, 256, 0, stream>>>(trg_tokens, trg_emb, EMBT);

    // ---- encoder layer 0 ----
    zero_kernel<<<(B_ * 1024 + 255) / 256, 256, 0, stream>>>(FIN, B_ * 1024);
    zero_kernel<<<(B_ * H_ + 255) / 256, 256, 0, stream>>>(HA, B_ * H_);
    for (int t = 0; t < S_; ++t) {
        const float* hp = (t & 1) ? HB : HA;
        float*       ho = (t & 1) ? HA : HB;
        lstm_step_kernel<<<dim3(8, 8), 128, 0, stream>>>(
            XBF + (long)t * B_ * 512, hp, ho, FIN, 1024, packE0,
            enc_bih0, enc_bhh0, src_lens, t,
            YBF + (long)t * B_ * 512, nullptr, nullptr, 0);
    }

    // ---- encoder layer 1 ----
    zero_kernel<<<(B_ * H_ + 255) / 256, 256, 0, stream>>>(HA, B_ * H_);
    for (int t = 0; t < S_; ++t) {
        const float* hp = (t & 1) ? HB : HA;
        float*       ho = (t & 1) ? HA : HB;
        lstm_step_kernel<<<dim3(8, 8), 128, 0, stream>>>(
            YBF + (long)t * B_ * 512, hp, ho, FIN + 512, 1024, packE1,
            enc_bih1, enc_bhh1, src_lens, t,
            nullptr, YSF, nullptr, 0);
    }

    // ---- decoder init: c_dec = fincells @ W_init^T + b_init ; h_dec = tanh ----
    gemm_kernel<<<dim3(8, 8), 128, 0, stream>>>(FIN, 1024, packI, 32, b_init, CDEC, 512);
    tanh_kernel<<<(B_ * H_ + 255) / 256, 256, 0, stream>>>(CDEC, HDEC, B_ * H_);

    // ---- decoder LSTM step: A = [emb_t | h_dec], mask all-true ----
    float* hOut = dout + B_ * H_;                 // output #2 (h)
    lstm_step_kernel<<<dim3(8, 8), 128, 0, stream>>>(
        EMBT, HDEC, hOut, CDEC, 512, packD,
        dec_bih, dec_bhh, nullptr, 0,
        nullptr, nullptr, ACAT, 1024);            // ACAT[:, :512] = h

    // ---- attention: context -> d_out[0..] and ACAT[:, 512:] ----
    attention_kernel<<<B_, 256, 0, stream>>>(YSF, hOut, src_lens, dout, ACAT);

    // ---- attn_vec = [h|context] @ W_hid^T + b_hid ----
    gemm_kernel<<<dim3(8, 8), 128, 0, stream>>>(ACAT, 1024, packHid, 32, b_hid, ATTNV, 512);

    // ---- logits = attn_vec @ W_out^T + b_out ; argmax ----
    gemm_kernel<<<dim3(500, 8), 128, 0, stream>>>(ATTNV, 512, packOut, 16, b_out, LOGITS, V_);
    argmax_kernel<<<B_, 256, 0, stream>>>(LOGITS, dout + 2 * B_ * H_);
}
